// NetworkCTR_Old_498216206935
// MI455X (gfx1250) — compile-verified, matrix operand order checked by reading, not run on hardware
//
#include <hip/hip_runtime.h>
#include <hip/hip_bf16.h>
#include <math.h>

#define NUM_FIELDS 24
#define EMBED_DIM  64
#define BATCH      4096
#define NUM_PAIRS  276
#define VOCAB      100000

typedef _Float16 h2   __attribute__((ext_vector_type(2)));
typedef _Float16 h8   __attribute__((ext_vector_type(8)));
typedef _Float16 v16h __attribute__((ext_vector_type(16)));
typedef float    v8f  __attribute__((ext_vector_type(8)));

// Pack a 16x32 f16 A/B fragment from an LDS-resident 32x64 tile, following the
// CDNA5 16-bit A-matrix layout: lanes 0-15 hold row M=lane with K={k0..k0+7} and
// {k0+16..k0+23}; lanes 16-31 hold the same rows with K={k0+8..15},{k0+24..31}.
__device__ inline v16h frag_from_lds(const _Float16* Ew, int rowBase, int lane, int k0) {
  int r  = rowBase + (lane & 15);
  int kb = k0 + ((lane & 16) ? 8 : 0);
  const _Float16* p = Ew + r * EMBED_DIM + kb;
  h8 lo = *(const h8*)(p);        // 16-byte aligned ds_load_b128
  h8 hi = *(const h8*)(p + 16);
  v16h out;
  #pragma unroll
  for (int t = 0; t < 8; ++t) { out[t] = lo[t]; out[t + 8] = hi[t]; }
  return out;
}

__global__ __launch_bounds__(256)
void ctr_nasfm_wmma_kernel(const int* __restrict__ x,
                           const float* __restrict__ emb,
                           const float* __restrict__ lin,
                           const float* __restrict__ bias,
                           const float* __restrict__ geno,
                           float* __restrict__ out) {
  __shared__ float    Sg[32 * 32];                 // pair-weight matrix, upper tri
  __shared__ _Float16 Elds[8 * 32 * EMBED_DIM];    // per-wave 32x64 f16 tile

  const int tid  = threadIdx.x;
  const int lane = tid & 31;
  const int wave = tid >> 5;
  const int b    = blockIdx.x * 8 + wave;          // one wave32 per sample

  // ---- build S[i][j] = genotype[pair(i,j)] for i<j, zero elsewhere ----
  for (int t = tid; t < 32 * 32; t += 256) Sg[t] = 0.f;
  __syncthreads();
  for (int p = tid; p < NUM_PAIRS; p += 256) {
    int i = 0, rem = p;                            // np.triu_indices(24, k=1) order
    while (rem >= 23 - i) { rem -= 23 - i; ++i; }
    int j = i + 1 + rem;
    Sg[i * 32 + j] = geno[p];
  }

  // ---- gather: 32 lanes x float2 = one full 256B embedding row per request ----
  _Float16* Ew = &Elds[wave * 32 * EMBED_DIM];
  const int* xb = x + b * NUM_FIELDS;
  #pragma unroll 4
  for (int f = 0; f < NUM_FIELDS; ++f) {
    int xv = xb[f];
    size_t idx = (size_t)xv + (size_t)f * VOCAB;
    const float2* row = (const float2*)(emb + idx * (size_t)EMBED_DIM);
    float2 v = row[lane];
    h2 hv; hv.x = (_Float16)v.x; hv.y = (_Float16)v.y;
    *(h2*)(Ew + f * EMBED_DIM + 2 * lane) = hv;
  }
  #pragma unroll
  for (int f = NUM_FIELDS; f < 32; ++f) {          // zero-pad rows 24..31
    h2 z; z.x = (_Float16)0.f; z.y = (_Float16)0.f;
    *(h2*)(Ew + f * EMBED_DIM + 2 * lane) = z;
  }

  // ---- linear term: lane f gathers lin_table[idx_f] ----
  float part = 0.f;
  if (lane < NUM_FIELDS) {
    int xv = xb[lane];
    size_t idx = (size_t)xv + (size_t)lane * VOCAB;
    part = lin[idx];
  }

  __syncthreads();

  // ---- Gram = E * E^T via WMMA; only tiles touching the upper triangle ----
  v8f c00 = {}; v8f c01 = {}; v8f c11 = {};
  #pragma unroll
  for (int k0 = 0; k0 < EMBED_DIM; k0 += 32) {
    v16h a0 = frag_from_lds(Ew, 0,  lane, k0);
    v16h a1 = frag_from_lds(Ew, 16, lane, k0);
    c00 = __builtin_amdgcn_wmma_f32_16x16x32_f16(false, a0, false, a0, (short)0, c00, false, false);
    c01 = __builtin_amdgcn_wmma_f32_16x16x32_f16(false, a0, false, a1, (short)0, c01, false, false);
    c11 = __builtin_amdgcn_wmma_f32_16x16x32_f16(false, a1, false, a1, (short)0, c11, false, false);
  }

  // ---- weight by S and fold into the reduction ----
  // C-tile layout: VGPR r holds (M = r + 8*(lane>=16), N = lane&15)
  const int nlo = lane & 15;
  const int mo  = (lane & 16) ? 8 : 0;
  #pragma unroll
  for (int r = 0; r < 8; ++r) {
    int M = r + mo;
    part += c00[r] * Sg[M * 32 + nlo];               // tile (M 0..15,  N 0..15)
    part += c01[r] * Sg[M * 32 + 16 + nlo];          // tile (M 0..15,  N 16..31)
    part += c11[r] * Sg[(16 + M) * 32 + 16 + nlo];   // tile (M 16..31, N 16..31)
  }

  // wave32 tree reduction
  #pragma unroll
  for (int off = 16; off > 0; off >>= 1)
    part += __shfl_xor(part, off, 32);

  if (lane == 0) {
    float z = part + bias[0];
    out[b] = 1.f / (1.f + expf(-z));
  }
}

extern "C" void kernel_launch(void* const* d_in, const int* in_sizes, int n_in,
                              void* d_out, int out_size, void* d_ws, size_t ws_size,
                              hipStream_t stream) {
  (void)in_sizes; (void)n_in; (void)out_size; (void)d_ws; (void)ws_size;
  const int*   x    = (const int*)d_in[0];
  const float* emb  = (const float*)d_in[1];
  const float* lin  = (const float*)d_in[2];
  const float* bias = (const float*)d_in[3];
  const float* geno = (const float*)d_in[4];
  float* out = (float*)d_out;

  dim3 grid(BATCH / 8);   // 512 blocks, 8 wave32 per block, 1 sample per wave
  dim3 block(256);
  ctr_nasfm_wmma_kernel<<<grid, block, 0, stream>>>(x, emb, lin, bias, geno, out);
}
